// BoxMultiHeadedAttention_41575283425541
// MI455X (gfx1250) — matrix operand order repaired and checked by reading, not compiled, not run on hardware
//
#include <hip/hip_runtime.h>
#include <hip/hip_bf16.h>

// ---------------------------------------------------------------------------
// BoxMultiHeadedAttention for MI455X (gfx1250, wave32, WMMA).
// - All matmuls on v_wmma_f32_16x16x32_bf16 (f32 accumulate).
// - Geometry log-bias computed once per (b,i,j), shared across heads.
// - GEMM: double-buffered K-loop with GLOBAL_LOAD_ASYNC_TO_LDS staging
//   (ASYNCcnt) overlapping WMMA compute; falls back to sync copies if the
//   async builtins are unavailable.
// Workspace use: ~51 MB.
// ---------------------------------------------------------------------------

typedef __attribute__((ext_vector_type(16))) __bf16 v16bf;
typedef __attribute__((ext_vector_type(8)))  __bf16 v8bf;
typedef __attribute__((ext_vector_type(8)))  float  v8f;
typedef __attribute__((ext_vector_type(4)))  int    vi4;

#define HH     8
#define DK     64
#define DM     512
#define NN     512
#define BB     4
#define MROWS  (BB * NN)   // 2048

#if defined(__has_builtin)
# if __has_builtin(__builtin_amdgcn_global_load_async_to_lds_b128) && \
     __has_builtin(__builtin_amdgcn_s_wait_asynccnt)
#  define HAVE_ASYNC 1
# endif
#endif
#ifndef HAVE_ASYNC
# define HAVE_ASYNC 0
#endif

static __device__ __forceinline__ unsigned short f2bf(float f) {
    union { float f; unsigned int u; } v; v.f = f;
    unsigned int r = v.u + 0x7FFFu + ((v.u >> 16) & 1u);  // RNE
    return (unsigned short)(r >> 16);
}

// 16B global -> LDS copy; async (ASYNCcnt) when available.
static __device__ __forceinline__ void copy16(const unsigned short* g,
                                              unsigned short* l) {
#if HAVE_ASYNC
    typedef __attribute__((address_space(1))) void gvoid;
    typedef __attribute__((address_space(3))) void lvoid;
    typedef __attribute__((address_space(1))) vi4  gvi4;
    typedef __attribute__((address_space(3))) vi4  lvi4;
    __builtin_amdgcn_global_load_async_to_lds_b128(
        (gvi4*)(gvoid*)(void*)const_cast<unsigned short*>(g),
        (lvi4*)(lvoid*)(void*)l, 0, 0);
#else
    *(uint4*)l = *(const uint4*)g;
#endif
}

static __device__ __forceinline__ void async_wait0() {
#if HAVE_ASYNC
    __builtin_amdgcn_s_wait_asynccnt(0);
#endif
}

// Build a 16-element bf16 fragment from two 16B LDS loads.
static __device__ __forceinline__ v16bf load_frag16(const unsigned short* p0,
                                                    const unsigned short* p1) {
    v8bf lo = *(const v8bf*)p0;
    v8bf hi = *(const v8bf*)p1;
    return __builtin_shufflevector(lo, hi, 0,1,2,3,4,5,6,7,8,9,10,11,12,13,14,15);
}

static __device__ __forceinline__ v8f wmma_bf16(v16bf a, v16bf b, v8f c) {
    return __builtin_amdgcn_wmma_f32_16x16x32_bf16(false, a, false, b,
                                                   (short)0, c, false, false);
}

// ------------------------------ f32 -> bf16 --------------------------------
__global__ void cvt_bf16_kernel(const float* __restrict__ src,
                                unsigned short* __restrict__ dst, int n) {
    int i = blockIdx.x * 256 + threadIdx.x;
    if (i < n) dst[i] = f2bf(src[i]);
}

// ------------------------- per-box geometry features -----------------------
__global__ void boxfeat_kernel(const float* __restrict__ box,
                               float* __restrict__ gf) {
    int i = blockIdx.x * 256 + threadIdx.x;
    if (i >= BB * NN) return;
    float x0 = box[i * 4 + 0], y0 = box[i * 4 + 1];
    float x1 = box[i * 4 + 2], y1 = box[i * 4 + 3];
    float cx = (x0 + x1) * 0.5f, cy = (y0 + y1) * 0.5f;
    float w = (x1 - x0) + 1.0f,  h = (y1 - y0) + 1.0f;
    gf[i * 6 + 0] = cx; gf[i * 6 + 1] = cy;
    gf[i * 6 + 2] = w;  gf[i * 6 + 3] = h;
    gf[i * 6 + 4] = logf(w); gf[i * 6 + 5] = logf(h);
}

// --------------------- geometry log-bias, all heads at once ----------------
__global__ __launch_bounds__(256)
void geom_kernel(const float* __restrict__ gf, const float* __restrict__ WG,
                 const float* __restrict__ bG, float* __restrict__ wglog) {
    __shared__ float sWG[HH * 64];
    __shared__ float sbG[HH];
    int tid = threadIdx.x;
    if (tid < HH * 64) sWG[tid] = WG[tid];
    if (tid < HH)      sbG[tid] = bG[tid];
    __syncthreads();

    int idx = blockIdx.x * 256 + tid;          // over B*N*N
    int j = idx & (NN - 1);
    int i = (idx >> 9) & (NN - 1);
    int b = idx >> 18;

    const float* gi = &gf[(b * NN + i) * 6];
    const float* gj = &gf[(b * NN + j) * 6];
    float pos[4];
    pos[0] = logf(fmaxf(fabsf((gi[0] - gj[0]) / gi[2]), 1e-3f));
    pos[1] = logf(fmaxf(fabsf((gi[1] - gj[1]) / gi[3]), 1e-3f));
    pos[2] = gi[4] - gj[4];
    pos[3] = gi[5] - gj[5];

    const float C[8] = {100.0f, 42.169650f, 17.782794f, 7.4989421f,
                        3.1622777f, 1.3335214f, 0.56234133f, 0.23713737f};
    float acc[HH];
    #pragma unroll
    for (int h = 0; h < HH; ++h) acc[h] = sbG[h];

    #pragma unroll
    for (int p = 0; p < 4; ++p) {
        #pragma unroll
        for (int f = 0; f < 8; ++f) {
            float ang = pos[p] * C[f];
            float s = __sinf(ang), c = __cosf(ang);
            int g = p * 8 + f;
            #pragma unroll
            for (int h = 0; h < HH; ++h)
                acc[h] += s * sWG[h * 64 + g] + c * sWG[h * 64 + 32 + g];
        }
    }
    #pragma unroll
    for (int h = 0; h < HH; ++h) {
        float wg = fmaxf(acc[h], 0.0f);
        wglog[(((size_t)(b * HH + h) * NN) + i) * NN + j] =
            logf(fmaxf(wg, 1e-6f));
    }
}

// ------------------------------- WMMA GEMM ---------------------------------
// C[m,n] = sum_k A[m,k]*W[n,k] + bias[n]   (M=2048, N=512, K=512, bf16 in)
// Double-buffered LDS, async global->LDS staging overlapped with WMMA.
// mode 0: store bf16 head-major  Cb[((b*8+h)*512 + n_row)*64 + d]
// mode 1: store f32 row-major    Cf[m*512 + n]
#define TILE_E (128 * 72)                 // ushorts per tile buffer
#define SMEM_GEMM (4 * TILE_E * 2)        // 2 bufs x (As+Ws) = 73728 B

__global__ __launch_bounds__(256)
void gemm_bf16_kernel(const unsigned short* __restrict__ A,
                      const unsigned short* __restrict__ W,
                      const float* __restrict__ bias,
                      float* __restrict__ Cf, unsigned short* __restrict__ Cb,
                      int mode) {
    extern __shared__ char gsm[];
    unsigned short* AsB = (unsigned short*)gsm;       // [2][128*72]
    unsigned short* WsB = AsB + 2 * TILE_E;           // [2][128*72]

    int tid  = threadIdx.x;
    int lane = tid & 31;
    int w    = tid >> 5;
    int wm   = w & 3;        // 32-row band
    int wn   = w >> 2;       // 64-col band
    int m0   = blockIdx.x * 128;
    int n0   = blockIdx.y * 128;
    int lh   = lane >> 4, ln = lane & 15;

    v8f acc[2][4];
    const v8f z = {0.f,0.f,0.f,0.f,0.f,0.f,0.f,0.f};
    #pragma unroll
    for (int a = 0; a < 2; ++a)
        #pragma unroll
        for (int c = 0; c < 4; ++c) acc[a][c] = z;

    auto stage = [&](int buf, int kk) {
        #pragma unroll
        for (int p = 0; p < 4; ++p) {
            int idx = tid + p * 256;
            int row = idx >> 3, q = idx & 7;
            copy16(&A[(size_t)(m0 + row) * DM + kk + q * 8],
                   &AsB[buf * TILE_E + row * 72 + q * 8]);
            copy16(&W[(size_t)(n0 + row) * DM + kk + q * 8],
                   &WsB[buf * TILE_E + row * 72 + q * 8]);
        }
    };

    stage(0, 0);
    async_wait0();
    __syncthreads();

    int cur = 0;
    for (int kk = 0; kk < DM; kk += 64) {
        if (kk + 64 < DM) stage(cur ^ 1, kk + 64);   // prefetch next tile

        const unsigned short* As = AsB + cur * TILE_E;
        const unsigned short* Ws = WsB + cur * TILE_E;
        #pragma unroll
        for (int ks = 0; ks < 2; ++ks) {
            int kb  = ks * 32 + lh * 16;   // A fragment K base
            int kbb = ks * 32 + lh * 16;   // B fragment K base
            v16bf af[2];
            #pragma unroll
            for (int tr = 0; tr < 2; ++tr) {
                int row = wm * 32 + tr * 16 + ln;
                af[tr] = load_frag16(&As[row * 72 + kb], &As[row * 72 + kb + 8]);
            }
            #pragma unroll
            for (int tc = 0; tc < 4; ++tc) {
                int wrow = wn * 64 + tc * 16 + ln;
                v16bf bf_ = load_frag16(&Ws[wrow * 72 + kbb],
                                        &Ws[wrow * 72 + kbb + 8]);
                #pragma unroll
                for (int tr = 0; tr < 2; ++tr)
                    acc[tr][tc] = wmma_bf16(af[tr], bf_, acc[tr][tc]);
            }
        }
        async_wait0();
        __syncthreads();
        cur ^= 1;
    }

    #pragma unroll
    for (int tr = 0; tr < 2; ++tr)
        #pragma unroll
        for (int tc = 0; tc < 4; ++tc)
            #pragma unroll
            for (int r = 0; r < 8; ++r) {
                int m = m0 + wm * 32 + tr * 16 + r + 8 * lh;
                int n = n0 + wn * 64 + tc * 16 + ln;
                float v = acc[tr][tc][r] + bias[n];
                if (mode == 0) {
                    int b = m >> 9, nr = m & (NN - 1);
                    int h = n >> 6, d  = n & (DK - 1);
                    Cb[(((size_t)(b * HH + h) * NN) + nr) * DK + d] = f2bf(v);
                } else {
                    Cf[(size_t)m * DM + n] = v;
                }
            }
}

// ------------------------------- attention ---------------------------------
#define LOGP 516
#define PBP  520
#define SMEM_ATTN (64*72*2*3 + 64*PBP*2 + 64*LOGP*4 + 64*8*4 + 64*4) // 228608 B

__global__ __launch_bounds__(256)
void attn_kernel(const unsigned short* __restrict__ Qh,
                 const unsigned short* __restrict__ Kh,
                 const unsigned short* __restrict__ Vh,
                 const float* __restrict__ wglog,
                 unsigned short* __restrict__ AO) {
    extern __shared__ char smem[];
    unsigned short* qs     = (unsigned short*)smem;          // [64][72]
    unsigned short* ks     = qs  + 64 * 72;                  // [64][72]
    unsigned short* vsT    = ks  + 64 * 72;                  // [64][72] (d-major)
    unsigned short* pb     = vsT + 64 * 72;                  // [64][520] bf16 P
    float* logits          = (float*)(pb + 64 * PBP);        // [64][516]
    float* red             = logits + 64 * LOGP;             // [64][8]
    float* rowinv          = red + 64 * 8;                   // [64]

    int tid  = threadIdx.x;
    int lane = tid & 31;
    int w    = tid >> 5;
    int tr   = w & 3;              // 16-row tile within 64
    int tc0  = (w >> 2) * 2;       // first of two 16-col tiles
    int lh   = lane >> 4, ln = lane & 15;

    int bid = blockIdx.x;
    int it  = bid & 7;
    int h   = (bid >> 3) & 7;
    int b   = bid >> 6;
    int i0  = it * 64;
    size_t bh = (size_t)(b * HH + h);

    // ---- stage Q tile (async)
    #pragma unroll
    for (int p = 0; p < 2; ++p) {
        int idx = tid + p * 256;
        int row = idx >> 3, q = idx & 7;
        copy16(&Qh[(bh * NN + i0 + row) * DK + q * 8], &qs[row * 72 + q * 8]);
    }

    // ---- phase 1: scores * scale + log-bias -> logits
    for (int jc = 0; jc < 8; ++jc) {
        int j0 = jc * 64;
        __syncthreads();
        #pragma unroll
        for (int p = 0; p < 2; ++p) {
            int idx = tid + p * 256;
            int row = idx >> 3, q = idx & 7;
            copy16(&Kh[(bh * NN + j0 + row) * DK + q * 8],
                   &ks[row * 72 + q * 8]);
        }
        async_wait0();
        __syncthreads();

        v8f sac[2];
        const v8f z = {0.f,0.f,0.f,0.f,0.f,0.f,0.f,0.f};
        sac[0] = z; sac[1] = z;
        #pragma unroll
        for (int kd = 0; kd < 2; ++kd) {
            int kb  = kd * 32 + lh * 16;
            int kbb = kd * 32 + lh * 16;
            int qrow = tr * 16 + ln;
            v16bf af = load_frag16(&qs[qrow * 72 + kb], &qs[qrow * 72 + kb + 8]);
            #pragma unroll
            for (int t = 0; t < 2; ++t) {
                int krow = (tc0 + t) * 16 + ln;
                v16bf bfr = load_frag16(&ks[krow * 72 + kbb],
                                        &ks[krow * 72 + kbb + 8]);
                sac[t] = wmma_bf16(af, bfr, sac[t]);
            }
        }
        #pragma unroll
        for (int t = 0; t < 2; ++t)
            #pragma unroll
            for (int r = 0; r < 8; ++r) {
                int il = tr * 16 + r + 8 * lh;
                int jl = (tc0 + t) * 16 + ln;
                float sc = sac[t][r] * 0.125f +
                    wglog[(bh * NN + i0 + il) * NN + j0 + jl];
                logits[il * LOGP + j0 + jl] = sc;
            }
    }

    // ---- softmax over each 512-wide row (4 threads per row)
    __syncthreads();
    {
        int row = tid >> 2, qt = tid & 3, base = qt * 128;
        float mx = -3.0e38f;
        for (int t = 0; t < 128; ++t)
            mx = fmaxf(mx, logits[row * LOGP + base + t]);
        red[row * 8 + qt] = mx;
        __syncthreads();
        float rm = fmaxf(fmaxf(red[row * 8 + 0], red[row * 8 + 1]),
                         fmaxf(red[row * 8 + 2], red[row * 8 + 3]));
        float s = 0.f;
        for (int t = 0; t < 128; ++t) {
            float e = __expf(logits[row * LOGP + base + t] - rm);
            pb[row * PBP + base + t] = f2bf(e);
            s += e;
        }
        red[row * 8 + 4 + qt] = s;
        __syncthreads();
        if (qt == 0) {
            float st = red[row * 8 + 4] + red[row * 8 + 5] +
                       red[row * 8 + 6] + red[row * 8 + 7];
            rowinv[row] = 1.0f / st;
        }
    }

    // ---- phase 2: O = P @ V (V staged transposed in LDS, d-major)
    v8f oc[2];
    {
        const v8f z = {0.f,0.f,0.f,0.f,0.f,0.f,0.f,0.f};
        oc[0] = z; oc[1] = z;
    }
    for (int jc = 0; jc < 8; ++jc) {
        int j0 = jc * 64;
        __syncthreads();
        #pragma unroll
        for (int p = 0; p < 2; ++p) {
            int idx = tid + p * 256;
            int jrow = idx >> 3, q = idx & 7;
            uint4 vv = *(const uint4*)&Vh[(bh * NN + j0 + jrow) * DK + q * 8];
            const unsigned short* ve = (const unsigned short*)&vv;
            #pragma unroll
            for (int e = 0; e < 8; ++e)
                vsT[(q * 8 + e) * 72 + jrow] = ve[e];
        }
        __syncthreads();
        #pragma unroll
        for (int kd = 0; kd < 2; ++kd) {
            int kb  = j0 + kd * 32 + lh * 16;
            int kbb = kd * 32 + lh * 16;
            int prow = tr * 16 + ln;
            v16bf af = load_frag16(&pb[prow * PBP + kb],
                                   &pb[prow * PBP + kb + 8]);
            #pragma unroll
            for (int t = 0; t < 2; ++t) {
                int drow = (tc0 + t) * 16 + ln;
                v16bf bfr = load_frag16(&vsT[drow * 72 + kbb],
                                        &vsT[drow * 72 + kbb + 8]);
                oc[t] = wmma_bf16(af, bfr, oc[t]);
            }
        }
    }

    // ---- normalize, store bf16 concat-head layout [b, n, h*64+d]
    #pragma unroll
    for (int t = 0; t < 2; ++t)
        #pragma unroll
        for (int r = 0; r < 8; ++r) {
            int il = tr * 16 + r + 8 * lh;
            int d  = (tc0 + t) * 16 + ln;
            float v = oc[t][r] * rowinv[il];
            AO[((size_t)(b * NN) + i0 + il) * DM + h * DK + d] = f2bf(v);
        }
}

// ------------------------------- launcher ----------------------------------
extern "C" void kernel_launch(void* const* d_in, const int* in_sizes, int n_in,
                              void* d_out, int out_size, void* d_ws, size_t ws_size,
                              hipStream_t stream) {
    (void)in_sizes; (void)n_in; (void)out_size; (void)ws_size;
    const float* Xq  = (const float*)d_in[0];
    const float* Xk  = (const float*)d_in[1];
    const float* Xv  = (const float*)d_in[2];
    const float* box = (const float*)d_in[3];
    const float* Wq  = (const float*)d_in[4];
    const float* bq  = (const float*)d_in[5];
    const float* Wk  = (const float*)d_in[6];
    const float* bk  = (const float*)d_in[7];
    const float* Wv  = (const float*)d_in[8];
    const float* bv  = (const float*)d_in[9];
    const float* Wo  = (const float*)d_in[10];
    const float* bo  = (const float*)d_in[11];
    const float* WG  = (const float*)d_in[12];
    const float* bG  = (const float*)d_in[13];
    float* out = (float*)d_out;

    char* ws = (char*)d_ws;                       // ~51 MB used
    unsigned short* Xqb = (unsigned short*)(ws + 0);
    unsigned short* Xkb = (unsigned short*)(ws + 2097152);
    unsigned short* Xvb = (unsigned short*)(ws + 4194304);
    unsigned short* Wqb = (unsigned short*)(ws + 6291456);
    unsigned short* Wkb = (unsigned short*)(ws + 6815744);
    unsigned short* Wvb = (unsigned short*)(ws + 7340032);
    unsigned short* Wob = (unsigned short*)(ws + 7864320);
    unsigned short* Qh  = (unsigned short*)(ws + 8388608);
    unsigned short* Kh  = (unsigned short*)(ws + 10485760);
    unsigned short* Vh  = (unsigned short*)(ws + 12582912);
    unsigned short* AO  = (unsigned short*)(ws + 14680064);
    float* gf           = (float*)(ws + 16777216);
    float* wglog        = (float*)(ws + 16826368);

    const int NX = MROWS * DM;   // 1048576
    const int NW = DM * DM;      //  262144

    cvt_bf16_kernel<<<NX / 256, 256, 0, stream>>>(Xq, Xqb, NX);
    cvt_bf16_kernel<<<NX / 256, 256, 0, stream>>>(Xk, Xkb, NX);
    cvt_bf16_kernel<<<NX / 256, 256, 0, stream>>>(Xv, Xvb, NX);
    cvt_bf16_kernel<<<NW / 256, 256, 0, stream>>>(Wq, Wqb, NW);
    cvt_bf16_kernel<<<NW / 256, 256, 0, stream>>>(Wk, Wkb, NW);
    cvt_bf16_kernel<<<NW / 256, 256, 0, stream>>>(Wv, Wvb, NW);
    cvt_bf16_kernel<<<NW / 256, 256, 0, stream>>>(Wo, Wob, NW);

    boxfeat_kernel<<<(BB * NN) / 256, 256, 0, stream>>>(box, gf);
    geom_kernel<<<(BB * NN * NN) / 256, 256, 0, stream>>>(gf, WG, bG, wglog);

    dim3 gg(MROWS / 128, DM / 128);
    gemm_bf16_kernel<<<gg, 256, SMEM_GEMM, stream>>>(Xqb, Wqb, bq, nullptr, Qh, 0);
    gemm_bf16_kernel<<<gg, 256, SMEM_GEMM, stream>>>(Xkb, Wkb, bk, nullptr, Kh, 0);
    gemm_bf16_kernel<<<gg, 256, SMEM_GEMM, stream>>>(Xvb, Wvb, bv, nullptr, Vh, 0);

    attn_kernel<<<BB * HH * (NN / 64), 256, SMEM_ATTN, stream>>>(
        Qh, Kh, Vh, wglog, AO);

    gemm_bf16_kernel<<<gg, 256, SMEM_GEMM, stream>>>(AO, Wob, bo, out, nullptr, 1);
}
